// SACCritic_77498389889733
// MI455X (gfx1250) — compile-verified, matrix operand order checked by reading, not run on hardware
//
#include <hip/hip_runtime.h>
#include <math.h>

// ---------------------------------------------------------------------------
// SAC critic for MI455X (gfx1250, wave32, WMMA).
//   K1: fused sLSTM scan  (A=[h|x] 16x192 @ W 192x512 per step, bf16 WMMA,
//       W_hh pre-swizzled B-fragments in LDS, W_ih fragments in VGPRs,
//       gate-interleaved columns so c/n/h update is per-lane VALU)
//   K2: k/v projection GEMM (seq_h @ Wk,Wv), async-to-LDS staging
//   K3: last-query attention (online softmax, VALU dot + shfl reduce)
//       + Wo + twin MLP heads (WMMA for all 128-wide GEMMs)
// Workspace needed: 3 * B*S*H * 2 bytes = 192 MiB (seq_h, k_all, v_all bf16).
// ---------------------------------------------------------------------------

#define BB   1024
#define SS   256
#define DD   64
#define HH   128
#define GG   512      // 4*H
#define NHEAD 4
#define DH   32

typedef unsigned short u16;
typedef unsigned int   u32;
typedef __attribute__((ext_vector_type(16))) __bf16 v16bf;
typedef __attribute__((ext_vector_type(8)))  float  v8f;

union BF16x16 { u16 u[16]; v16bf v; uint4 q[2]; };

__device__ inline u16 f2bf(float f) {
    union { float f; u32 u; } x; x.f = f;
    u32 u = x.u;
    return (u16)((u + 0x7FFFu + ((u >> 16) & 1u)) >> 16);  // RNE
}
__device__ inline float bf2f(u16 h) {
    union { u32 u; float f; } x; x.u = ((u32)h) << 16;
    return x.f;
}

__device__ inline v8f wmma_bf16(v16bf a, v16bf b, v8f c) {
    return __builtin_amdgcn_wmma_f32_16x16x32_bf16(false, a, false, b, (short)0, c, false, false);
}

// Gather a B-matrix fragment (32K x 16N tile) from a row-major fp32 weight
// matrix, converting to bf16, laid out per the CDNA5 16-bit B VGPR layout:
// lane n<16 holds column colBase+n, K = ktBase+{0..7,16..23}; lanes>=16 K+=8.
// VGPR j packs K pair (2j, 2j+1) with the d>=4 dwords holding K 16..23.
__device__ inline v16bf gather_bfrag(const float* __restrict__ W, int ldn,
                                     int colBase, int ktBase, int Kmax) {
    const int lane = threadIdx.x & 31;
    const int c = colBase + (lane & 15);
    const int khalf = (lane & 16) ? 8 : 0;
    BF16x16 r;
#pragma unroll
    for (int d = 0; d < 8; ++d) {
        int k = ktBase + khalf + 2 * d + ((d >= 4) ? 8 : 0);
        float f0 = (k     < Kmax) ? W[(size_t)k * ldn + c]       : 0.f;
        float f1 = (k + 1 < Kmax) ? W[(size_t)(k + 1) * ldn + c] : 0.f;
        r.u[2 * d]     = f2bf(f0);
        r.u[2 * d + 1] = f2bf(f1);
    }
    return r.v;
}

// A-matrix fragment (16M x 32K) from row-major bf16 LDS buffer (stride in halves).
// lane holds row rowBase+(lane&15); two contiguous 16B chunks: K k0..k0+7 and k0+16..k0+23.
__device__ inline v16bf afrag_lds(const u16* buf, int stride, int rowBase, int ktBase) {
    const int lane = threadIdx.x & 31;
    const int r  = rowBase + (lane & 15);
    const int k0 = ktBase + ((lane & 16) ? 8 : 0);
    const u16* p = buf + r * stride + k0;
    BF16x16 t;
    t.q[0] = *(const uint4*)(p);
    t.q[1] = *(const uint4*)(p + 16);
    return t.v;
}

// A-fragment straight from global fp32 x (B,S,D): per-lane cvt to bf16.
__device__ inline v16bf afrag_x(const float* __restrict__ x, int b0, int t, int ktBase) {
    const int lane = threadIdx.x & 31;
    const float* p = x + (size_t)(b0 + (lane & 15)) * (SS * DD) + (size_t)t * DD
                       + ktBase + ((lane & 16) ? 8 : 0);
    float4 a0 = *(const float4*)(p);
    float4 a1 = *(const float4*)(p + 4);
    float4 b0v = *(const float4*)(p + 16);
    float4 b1v = *(const float4*)(p + 20);
    BF16x16 r;
    r.u[0]  = f2bf(a0.x); r.u[1]  = f2bf(a0.y); r.u[2]  = f2bf(a0.z); r.u[3]  = f2bf(a0.w);
    r.u[4]  = f2bf(a1.x); r.u[5]  = f2bf(a1.y); r.u[6]  = f2bf(a1.z); r.u[7]  = f2bf(a1.w);
    r.u[8]  = f2bf(b0v.x); r.u[9]  = f2bf(b0v.y); r.u[10] = f2bf(b0v.z); r.u[11] = f2bf(b0v.w);
    r.u[12] = f2bf(b1v.x); r.u[13] = f2bf(b1v.y); r.u[14] = f2bf(b1v.z); r.u[15] = f2bf(b1v.w);
    return r.v;
}

// ---------------------------------------------------------------------------
// Kernel 1: sLSTM scan. 64 blocks (16 batch rows each), 8 waves.
// Wave w owns hidden units [16w,16w+16) for all 4 gates (column permutation:
// gate g, unit block w -> original col g*128 + 16w + c).
// ---------------------------------------------------------------------------
__global__ void __launch_bounds__(256)
k_scan(const float* __restrict__ x, const float* __restrict__ W_ih,
       const float* __restrict__ W_hh, const float* __restrict__ b_cell,
       u16* __restrict__ seq_h) {
    __shared__ u16 whhFrag[4 * 4 * 8 * 32 * 16];  // [g][kt][w][lane][16] = 128 KiB
    __shared__ u16 hbuf[2][16 * 136];             // double-buffered h (padded stride)

    const int tid  = threadIdx.x;
    const int w    = tid >> 5;
    const int lane = tid & 31;
    const int b0   = blockIdx.x * 16;

    // ---- one-time: pre-swizzle W_hh into B-fragment layout in LDS ----
    for (int idx = tid; idx < 4 * 4 * 8 * 32 * 8; idx += 256) {
        int d  = idx & 7;
        int l  = (idx >> 3) & 31;
        int wv = (idx >> 8) & 7;
        int kt = (idx >> 11) & 3;
        int g  = idx >> 13;
        int k   = kt * 32 + ((l & 16) ? 8 : 0) + 2 * d + ((d >= 4) ? 8 : 0);
        int col = g * 128 + wv * 16 + (l & 15);
        int base = (((g * 4 + kt) * 8 + wv) * 32 + l) * 16 + 2 * d;
        whhFrag[base]     = f2bf(W_hh[(size_t)k * GG + col]);
        whhFrag[base + 1] = f2bf(W_hh[(size_t)(k + 1) * GG + col]);
    }
    // ---- W_ih fragments in VGPRs (per wave: 4 gates x 2 k-tiles) ----
    v16bf wih[4][2];
#pragma unroll
    for (int g = 0; g < 4; ++g)
#pragma unroll
        for (int kt = 0; kt < 2; ++kt)
            wih[g][kt] = gather_bfrag(W_ih, GG, g * 128 + w * 16, kt * 32, DD);

    float bias[4];
#pragma unroll
    for (int g = 0; g < 4; ++g) bias[g] = b_cell[g * 128 + w * 16 + (lane & 15)];

    for (int i = tid; i < 16 * 136; i += 256) hbuf[0][i] = 0;  // h_{-1} = 0

    v8f c = {}, n;
#pragma unroll
    for (int j = 0; j < 8; ++j) n[j] = 1.f;
    __syncthreads();

    for (int t = 0; t < SS; ++t) {
        const int cur = t & 1, nxt = cur ^ 1;
        // stream h_{t-1} to HBM (coalesced b128 from LDS)
        if (t > 0) {
            int row = tid >> 4, ch = tid & 15;
            uint4 v = *(const uint4*)&hbuf[cur][row * 136 + ch * 8];
            *(uint4*)(seq_h + ((size_t)(b0 + row) * SS + (t - 1)) * HH + ch * 8) = v;
        }
        v8f acc[4];
#pragma unroll
        for (int g = 0; g < 4; ++g)
#pragma unroll
            for (int j = 0; j < 8; ++j) acc[g][j] = bias[g];

        // h @ W_hh  (K = 128)
#pragma unroll
        for (int kt = 0; kt < 4; ++kt) {
            v16bf a = afrag_lds(hbuf[cur], 136, 0, kt * 32);
#pragma unroll
            for (int g = 0; g < 4; ++g) {
                const u16* fp = &whhFrag[(((g * 4 + kt) * 8 + w) * 32 + lane) * 16];
                BF16x16 bb;
                bb.q[0] = *(const uint4*)fp;
                bb.q[1] = *(const uint4*)(fp + 8);
                acc[g] = wmma_bf16(a, bb.v, acc[g]);
            }
        }
        // x_t @ W_ih  (K = 64)
#pragma unroll
        for (int kt = 0; kt < 2; ++kt) {
            v16bf a = afrag_x(x, b0, t, kt * 32);
#pragma unroll
            for (int g = 0; g < 4; ++g) acc[g] = wmma_bf16(a, wih[g][kt], acc[g]);
        }
        // sLSTM state update: pure per-lane VALU (tile layouts line up)
        v8f hnew;
#pragma unroll
        for (int j = 0; j < 8; ++j) {
            float iv = expf(acc[0][j]);
            float fv = 1.f / (1.f + expf(-acc[1][j]));
            float zv = tanhf(acc[2][j]);
            float ov = 1.f / (1.f + expf(-acc[3][j]));
            float cv = fv * c[j] + iv * zv;
            float nv = fv * n[j] + iv;
            c[j] = cv; n[j] = nv;
            hnew[j] = ov * (cv / nv);
        }
        const int colh  = w * 16 + (lane & 15);
        const int rbase = (lane & 16) ? 8 : 0;
#pragma unroll
        for (int j = 0; j < 8; ++j)
            hbuf[nxt][(rbase + j) * 136 + colh] = f2bf(hnew[j]);
        __syncthreads();
    }
    {   // final h_{S-1}
        int row = tid >> 4, ch = tid & 15;
        uint4 v = *(const uint4*)&hbuf[SS & 1][row * 136 + ch * 8];
        *(uint4*)(seq_h + ((size_t)(b0 + row) * SS + (SS - 1)) * HH + ch * 8) = v;
    }
}

// ---------------------------------------------------------------------------
// Kernel 2: k_all, v_all = seq_h @ Wk, Wv.  4096 blocks, 64 rows each.
// seq_h staged to LDS with gfx1250 async-to-LDS loads.
// ---------------------------------------------------------------------------
__global__ void __launch_bounds__(256)
k_kvproj(const u16* __restrict__ seq_h, const float* __restrict__ Wk,
         const float* __restrict__ Wv, u16* __restrict__ k_all,
         u16* __restrict__ v_all) {
    __shared__ u16 As[64 * 136];
    __shared__ u16 Os[16 * 256];
    const int tid = threadIdx.x, w = tid >> 5, lane = tid & 31;
    const size_t row0 = (size_t)blockIdx.x * 64;

    // async DMA seq_h tile (64x128 bf16 = 16 KiB) into LDS
    for (int idx = tid; idx < 64 * 16; idx += 256) {
        int r = idx >> 4, ch = idx & 15;
        const u16* gp = seq_h + (row0 + r) * HH + ch * 8;
        u32 lds_off = (u32)(size_t)&As[r * 136 + ch * 8];
        asm volatile("global_load_async_to_lds_b128 %0, %1, off"
                     :: "v"(lds_off), "v"(gp) : "memory");
    }
    v16bf wkF[4], wvF[4];
#pragma unroll
    for (int kt = 0; kt < 4; ++kt) {
        wkF[kt] = gather_bfrag(Wk, HH, w * 16, kt * 32, HH);
        wvF[kt] = gather_bfrag(Wv, HH, w * 16, kt * 32, HH);
    }
    asm volatile("s_wait_asynccnt 0" ::: "memory");
    __syncthreads();

    for (int m = 0; m < 4; ++m) {
        v8f ak = {}, av = {};
#pragma unroll
        for (int kt = 0; kt < 4; ++kt) {
            v16bf a = afrag_lds(As, 136, m * 16, kt * 32);
            ak = wmma_bf16(a, wkF[kt], ak);
            av = wmma_bf16(a, wvF[kt], av);
        }
        const int col = w * 16 + (lane & 15);
        const int rb  = (lane & 16) ? 8 : 0;
#pragma unroll
        for (int j = 0; j < 8; ++j) {
            Os[(rb + j) * 256 + col]       = f2bf(ak[j]);
            Os[(rb + j) * 256 + 128 + col] = f2bf(av[j]);
        }
        __syncthreads();
        for (int idx = tid; idx < 16 * 32; idx += 256) {
            int r = idx >> 5, ch = idx & 31;
            uint4 v = *(const uint4*)&Os[r * 256 + ch * 8];
            size_t grow = row0 + m * 16 + r;
            if (ch < 16) *(uint4*)(k_all + grow * HH + ch * 8)        = v;
            else         *(uint4*)(v_all + grow * HH + (ch - 16) * 8) = v;
        }
        __syncthreads();
    }
}

// ---------------------------------------------------------------------------
// Kernel 3: last-query attention + Wo + twin MLP heads. 64 blocks, 16 rows.
// ---------------------------------------------------------------------------
__global__ void __launch_bounds__(256)
k_attn_mlp(const u16* __restrict__ seq_h, const u16* __restrict__ k_all,
           const u16* __restrict__ v_all, const float* __restrict__ action,
           const float* __restrict__ info, const float* __restrict__ Wq,
           const float* __restrict__ Wo, const float* __restrict__ q_w1,
           const float* __restrict__ q_b1, const float* __restrict__ q_gamma,
           const float* __restrict__ q_beta, const float* __restrict__ q_w2,
           const float* __restrict__ q_b2, const float* __restrict__ q_w3,
           const float* __restrict__ q_b3, float* __restrict__ out) {
    __shared__ u16   Abuf[16 * 136];
    __shared__ float qs[16 * 128];
    __shared__ float fbuf[16 * 128];
    __shared__ u16   catb[16 * 168];   // K padded to 160 (stride 168)
    __shared__ u16   Ab2[16 * 136];
    __shared__ float rowstat[2][16];

    const int tid = threadIdx.x, w = tid >> 5, lane = tid & 31;
    const int b0 = blockIdx.x * 16;

    {   // h_last -> Abuf
        int r = tid >> 4, ch = tid & 15;
        uint4 v = *(const uint4*)(seq_h + ((size_t)(b0 + r) * SS + (SS - 1)) * HH + ch * 8);
        *(uint4*)&Abuf[r * 136 + ch * 8] = v;
    }
    // cat tail: info(13) | action(3) | zeros
    for (int idx = tid; idx < 16 * 40; idx += 256) {
        int r = idx / 40, cc = 128 + idx % 40;
        float v = 0.f;
        if (cc < 141)      v = info[(b0 + r) * 13 + (cc - 128)];
        else if (cc < 144) v = action[(b0 + r) * 3 + (cc - 141)];
        catb[r * 168 + cc] = f2bf(v);
    }
    v16bf wqF[4];
#pragma unroll
    for (int kt = 0; kt < 4; ++kt) wqF[kt] = gather_bfrag(Wq, HH, w * 16, kt * 32, HH);
    __syncthreads();

    {   // q = h_last @ Wq  (scale 1/sqrt(DH) folded in)
        v8f a = {};
#pragma unroll
        for (int kt = 0; kt < 4; ++kt) a = wmma_bf16(afrag_lds(Abuf, 136, 0, kt * 32), wqF[kt], a);
        const int col = w * 16 + (lane & 15), rb = (lane & 16) ? 8 : 0;
        const float sc = 0.17677669529663687f;
#pragma unroll
        for (int j = 0; j < 8; ++j) qs[(rb + j) * 128 + col] = a[j] * sc;
    }
    __syncthreads();

    {   // online-softmax attention over t: 4 lanes per (row, head)
        const int p = tid >> 2, row = p & 15, head = p >> 4, sub = tid & 3;
        const int d0 = head * 32 + sub * 8;
        float qv[8];
#pragma unroll
        for (int j = 0; j < 8; ++j) qv[j] = qs[row * 128 + d0 + j];
        float m = -3.4e38f, l = 0.f, o[8];
#pragma unroll
        for (int j = 0; j < 8; ++j) o[j] = 0.f;
        const u16* kp = k_all + ((size_t)(b0 + row) * SS) * HH + d0;
        const u16* vp = v_all + ((size_t)(b0 + row) * SS) * HH + d0;
        for (int t = 0; t < SS; ++t) {
            uint4 kq4 = *(const uint4*)(kp + (size_t)t * HH);
            const u16* kh = (const u16*)&kq4;
            float s = 0.f;
#pragma unroll
            for (int j = 0; j < 8; ++j) s += qv[j] * bf2f(kh[j]);
            s += __shfl_xor(s, 1, 4);
            s += __shfl_xor(s, 2, 4);
            float mn = fmaxf(m, s);
            float scale = expf(m - mn);
            float pe = expf(s - mn);
            uint4 vq4 = *(const uint4*)(vp + (size_t)t * HH);
            const u16* vh = (const u16*)&vq4;
            l = l * scale + pe;
#pragma unroll
            for (int j = 0; j < 8; ++j) o[j] = o[j] * scale + pe * bf2f(vh[j]);
            m = mn;
        }
        float inv = 1.f / l;
#pragma unroll
        for (int j = 0; j < 8; ++j) fbuf[row * 128 + d0 + j] = o[j] * inv;
    }
    __syncthreads();

    // att -> bf16; context = att @ Wo -> catb[:,0:128]
    for (int idx = tid; idx < 2048; idx += 256) {
        int r = idx >> 7, cc = idx & 127;
        Ab2[r * 136 + cc] = f2bf(fbuf[r * 128 + cc]);
    }
    v16bf woF[4];
#pragma unroll
    for (int kt = 0; kt < 4; ++kt) woF[kt] = gather_bfrag(Wo, HH, w * 16, kt * 32, HH);
    __syncthreads();
    {
        v8f a = {};
#pragma unroll
        for (int kt = 0; kt < 4; ++kt) a = wmma_bf16(afrag_lds(Ab2, 136, 0, kt * 32), woF[kt], a);
        const int col = w * 16 + (lane & 15), rb = (lane & 16) ? 8 : 0;
#pragma unroll
        for (int j = 0; j < 8; ++j) catb[(rb + j) * 168 + col] = f2bf(a[j]);
    }
    __syncthreads();

    // twin Q-heads
    for (int kq = 0; kq < 2; ++kq) {
        const float* w1 = q_w1 + (size_t)kq * 144 * 128;
        const float* w2 = q_w2 + (size_t)kq * 128 * 128;
        const float* w3 = q_w3 + (size_t)kq * 128;
        v16bf w1F[5];
#pragma unroll
        for (int kt = 0; kt < 5; ++kt) w1F[kt] = gather_bfrag(w1, 128, w * 16, kt * 32, 144);
        {   // h1 = cat @ w1 + b1
            v8f a = {};
#pragma unroll
            for (int kt = 0; kt < 5; ++kt) a = wmma_bf16(afrag_lds(catb, 168, 0, kt * 32), w1F[kt], a);
            const int col = w * 16 + (lane & 15), rb = (lane & 16) ? 8 : 0;
            float b1 = q_b1[kq * 128 + col];
#pragma unroll
            for (int j = 0; j < 8; ++j) fbuf[(rb + j) * 128 + col] = a[j] + b1;
        }
        __syncthreads();
        if (tid < 16) {   // LayerNorm stats per row
            float s = 0.f, ss = 0.f;
            for (int cc = 0; cc < 128; ++cc) { float v = fbuf[tid * 128 + cc]; s += v; ss += v * v; }
            float mu = s * (1.f / 128.f);
            float var = ss * (1.f / 128.f) - mu * mu;
            rowstat[0][tid] = mu;
            rowstat[1][tid] = rsqrtf(var + 1e-5f);
        }
        __syncthreads();
        for (int idx = tid; idx < 2048; idx += 256) {  // LN*gamma+beta, exact gelu
            int r = idx >> 7, cc = idx & 127;
            float v = (fbuf[r * 128 + cc] - rowstat[0][r]) * rowstat[1][r]
                        * q_gamma[kq * 128 + cc] + q_beta[kq * 128 + cc];
            float g = 0.5f * v * (1.f + erff(v * 0.70710678118654752f));
            Ab2[r * 136 + cc] = f2bf(g);
        }
        v16bf w2F[4];
#pragma unroll
        for (int kt = 0; kt < 4; ++kt) w2F[kt] = gather_bfrag(w2, 128, w * 16, kt * 32, 128);
        __syncthreads();
        {   // h2 = gelu(h1g @ w2 + b2) -> Abuf
            v8f a = {};
#pragma unroll
            for (int kt = 0; kt < 4; ++kt) a = wmma_bf16(afrag_lds(Ab2, 136, 0, kt * 32), w2F[kt], a);
            const int col = w * 16 + (lane & 15), rb = (lane & 16) ? 8 : 0;
            float b2 = q_b2[kq * 128 + col];
#pragma unroll
            for (int j = 0; j < 8; ++j) {
                float v = a[j] + b2;
                float g = 0.5f * v * (1.f + erff(v * 0.70710678118654752f));
                Abuf[(rb + j) * 136 + col] = f2bf(g);
            }
        }
        __syncthreads();
        {   // qv = h2 @ w3 + b3  (16-lane shfl reduction)
            int row = tid >> 4, ch = tid & 15;
            float s = 0.f;
#pragma unroll
            for (int j = 0; j < 8; ++j) s += bf2f(Abuf[row * 136 + ch * 8 + j]) * w3[ch * 8 + j];
            s += __shfl_xor(s, 1, 16);
            s += __shfl_xor(s, 2, 16);
            s += __shfl_xor(s, 4, 16);
            s += __shfl_xor(s, 8, 16);
            if (ch == 0) out[kq * BB + b0 + row] = s + q_b3[kq];
        }
        __syncthreads();
    }
}

// ---------------------------------------------------------------------------
extern "C" void kernel_launch(void* const* d_in, const int* in_sizes, int n_in,
                              void* d_out, int out_size, void* d_ws, size_t ws_size,
                              hipStream_t stream) {
    const float* x      = (const float*)d_in[0];
    const float* action = (const float*)d_in[1];
    const float* info   = (const float*)d_in[2];
    const float* W_ih   = (const float*)d_in[3];
    const float* W_hh   = (const float*)d_in[4];
    const float* b_cell = (const float*)d_in[5];
    const float* Wq     = (const float*)d_in[6];
    const float* Wk     = (const float*)d_in[7];
    const float* Wv     = (const float*)d_in[8];
    const float* Wo     = (const float*)d_in[9];
    const float* q_w1   = (const float*)d_in[10];
    const float* q_b1   = (const float*)d_in[11];
    const float* q_gamma= (const float*)d_in[12];
    const float* q_beta = (const float*)d_in[13];
    const float* q_w2   = (const float*)d_in[14];
    const float* q_b2   = (const float*)d_in[15];
    const float* q_w3   = (const float*)d_in[16];
    const float* q_b3   = (const float*)d_in[17];
    float* out = (float*)d_out;

    const size_t seqBytes = (size_t)BB * SS * HH * sizeof(u16);  // 64 MiB each
    char* ws = (char*)d_ws;
    u16* seq_h = (u16*)(ws);
    u16* k_all = (u16*)(ws + seqBytes);
    u16* v_all = (u16*)(ws + 2 * seqBytes);

    k_scan<<<dim3(BB / 16), dim3(256), 0, stream>>>(x, W_ih, W_hh, b_cell, seq_h);
    k_kvproj<<<dim3((BB * SS) / 64), dim3(256), 0, stream>>>(seq_h, Wk, Wv, k_all, v_all);
    k_attn_mlp<<<dim3(BB / 16), dim3(256), 0, stream>>>(seq_h, k_all, v_all, action, info,
                                                        Wq, Wo, q_w1, q_b1, q_gamma, q_beta,
                                                        q_w2, q_b2, q_w3, q_b3, out);
    (void)in_sizes; (void)n_in; (void)out_size; (void)ws_size;
}
